// GNN_37203006718477
// MI455X (gfx1250) — matrix operand order repaired and checked by reading, not compiled, not run on hardware
//
#include <hip/hip_runtime.h>
#include <hip/hip_bf16.h>
#include <math.h>
#include <stdint.h>

typedef __attribute__((ext_vector_type(16))) _Float16 v16h;
typedef __attribute__((ext_vector_type(8)))  float    v8f;

#define SLOPE 0.01f
#define SCAN_CHUNK 2048

__device__ __forceinline__ float leakyf(float v) { return v > 0.0f ? v : SLOPE * v; }
__device__ __forceinline__ float sigf(float v)   { return 1.0f / (1.0f + __expf(-v)); }

// ---------------------------------------------------------------------------
// Weight packing: W[K x 64] f32 row-major -> f16 fragments in the exact
// per-lane VGPR layout of the WMMA B operand (32x16 f16 tile per fragment).
// Fragment f = kbi*4 + nt holds 32 lanes x 16 halfs:  element j of lane l is
// B[k = kbi*32 + (l>>4)*8 + (j&7) + (j>>3)*16][n = nt*16 + (l&15)].
// ---------------------------------------------------------------------------
__global__ void k_pack_w(const float* __restrict__ W, int K, _Float16* __restrict__ dst) {
    int idx = blockIdx.x * blockDim.x + threadIdx.x;
    int total = K * 64;
    if (idx >= total) return;
    int f   = idx >> 9;         // fragment
    int rem = idx & 511;
    int l   = rem >> 4;         // lane
    int j   = rem & 15;         // element
    int kbi = f >> 2, nt = f & 3;
    int m = l & 15, half = l >> 4;
    int k = kbi * 32 + half * 8 + (j & 7) + (j >> 3) * 16;
    int n = nt * 16 + m;
    dst[idx] = (_Float16)W[(long)k * 64 + n];
}

// ---------------------------------------------------------------------------
// WMMA GEMM: C[M x 64] = act(A[M x K] @ W[K x 64] + bias [+ add]).
// A: f32 (float4 loads, cvt to f16).  Wp: pre-packed f16 fragments.
// One wave = 16-row tile x all 4 column tiles. act: 0 none, 1 leaky, 2 tanh.
// ---------------------------------------------------------------------------
__global__ void k_gemm_wmma(const float* __restrict__ A, int lda, int K, int M,
                            const _Float16* __restrict__ Wp,
                            const float* __restrict__ bias,
                            const float* __restrict__ add, int add_before,
                            int act, float* __restrict__ C, int ldc)
{
    const int lane = threadIdx.x & 31;
    const int tile = blockIdx.x * (blockDim.x >> 5) + (threadIdx.x >> 5);
    const int row0 = tile * 16;
    if (row0 >= M) return;                 // whole wave exits together
    const int m    = lane & 15;
    const int half = lane >> 4;
    const long row = row0 + m;

    v8f acc[4] = {};

    const int nkb = K >> 5;
    for (int kbi = 0; kbi < nkb; ++kbi) {
        // A fragment: elems 0..7 -> K=kb+half*8+j ; elems 8..15 -> +16
        const float4* Ar = (const float4*)(A + row * lda + kbi * 32 + half * 8);
        float4 a0 = Ar[0];
        float4 a1 = Ar[1];
        float4 a2 = Ar[4];   // +16 floats
        float4 a3 = Ar[5];
        v16h a;
        a[0]=(_Float16)a0.x; a[1]=(_Float16)a0.y; a[2]=(_Float16)a0.z; a[3]=(_Float16)a0.w;
        a[4]=(_Float16)a1.x; a[5]=(_Float16)a1.y; a[6]=(_Float16)a1.z; a[7]=(_Float16)a1.w;
        a[8]=(_Float16)a2.x; a[9]=(_Float16)a2.y; a[10]=(_Float16)a2.z; a[11]=(_Float16)a2.w;
        a[12]=(_Float16)a3.x; a[13]=(_Float16)a3.y; a[14]=(_Float16)a3.z; a[15]=(_Float16)a3.w;

        #pragma unroll
        for (int nt = 0; nt < 4; ++nt) {
            v16h b = *(const v16h*)(Wp + (((long)(kbi * 4 + nt)) << 9) + (lane << 4));
            acc[nt] = __builtin_amdgcn_wmma_f32_16x16x32_f16(
                false, a, false, b, (short)0, acc[nt], false, false);
        }
    }

    #pragma unroll
    for (int nt = 0; nt < 4; ++nt) {
        #pragma unroll
        for (int r = 0; r < 8; ++r) {
            long rr = row0 + half * 8 + r;
            int  cc = nt * 16 + m;
            float v = acc[nt][r];
            if (bias) v += bias[cc];
            if (add && add_before)  v += add[rr * 64 + cc];
            if (act == 1) v = leakyf(v);
            else if (act == 2) v = tanhf(v);
            if (add && !add_before) v += add[rr * 64 + cc];
            C[rr * ldc + cc] = v;
        }
    }
}

// ---------------------------------------------------------------------------
// Row-wise L2 normalize (optionally +bias first, optionally leaky after).
// ---------------------------------------------------------------------------
__global__ void k_rownorm(const float* __restrict__ src, int lds,
                          float* __restrict__ dst, int ldd,
                          const float* __restrict__ bias, int do_leaky, int Nn)
{
    int lane = threadIdx.x & 31;
    int r = blockIdx.x * (blockDim.x >> 5) + (threadIdx.x >> 5);
    if (r >= Nn) return;
    int d0 = lane * 2;
    float v0 = src[(long)r * lds + d0];
    float v1 = src[(long)r * lds + d0 + 1];
    if (bias) { v0 += bias[d0]; v1 += bias[d0 + 1]; }
    float ss = v0 * v0 + v1 * v1;
    #pragma unroll
    for (int off = 16; off > 0; off >>= 1) ss += __shfl_xor(ss, off, 32);
    float inv = 1.0f / fmaxf(sqrtf(ss), 1e-12f);
    v0 *= inv; v1 *= inv;
    if (do_leaky) { v0 = leakyf(v0); v1 = leakyf(v1); }
    dst[(long)r * ldd + d0]     = v0;
    dst[(long)r * ldd + d0 + 1] = v1;
}

// ------------------------------- small utils -------------------------------
__global__ void k_fillf(float* p, float v, int n) {
    int i = blockIdx.x * blockDim.x + threadIdx.x;
    if (i < n) p[i] = v;
}
__global__ void k_zeroi(int* p, int n) {
    int i = blockIdx.x * blockDim.x + threadIdx.x;
    if (i < n) p[i] = 0;
}
__global__ void k_seti(int* p, int v) { *p = v; }

__global__ void k_deg(const int* __restrict__ row, float* __restrict__ deg, int E) {
    int e = blockIdx.x * blockDim.x + threadIdx.x;
    if (e < E) atomicAdd(&deg[row[e]], 1.0f);
}
__global__ void k_dinv(float* d, int Nn) {
    int i = blockIdx.x * blockDim.x + threadIdx.x;
    if (i < Nn) { float v = d[i]; d[i] = v > 0.0f ? rsqrtf(v) : 0.0f; }
}

// ------------------------------- CSR build ---------------------------------
__global__ void k_count_col(const int* __restrict__ col, int* __restrict__ cnt, int E) {
    int e = blockIdx.x * blockDim.x + threadIdx.x;
    if (e < E) atomicAdd(&cnt[col[e]], 1);
}

// block-level exclusive scan over chunks of SCAN_CHUNK elems (Hillis-Steele in LDS)
__global__ void k_scan1(const int* __restrict__ in, int* __restrict__ out,
                        int* __restrict__ bsums, int n)
{
    __shared__ int sh[SCAN_CHUNK];
    int base = blockIdx.x * SCAN_CHUNK;
    for (int i = threadIdx.x; i < SCAN_CHUNK; i += blockDim.x)
        sh[i] = (base + i < n) ? in[base + i] : 0;
    __syncthreads();
    for (int off = 1; off < SCAN_CHUNK; off <<= 1) {
        int vals[SCAN_CHUNK / 256];
        for (int i = threadIdx.x, q = 0; i < SCAN_CHUNK; i += blockDim.x, ++q)
            vals[q] = (i >= off) ? sh[i - off] : 0;
        __syncthreads();
        for (int i = threadIdx.x, q = 0; i < SCAN_CHUNK; i += blockDim.x, ++q)
            sh[i] += vals[q];
        __syncthreads();
    }
    for (int i = threadIdx.x; i < SCAN_CHUNK; i += blockDim.x)
        if (base + i < n) out[base + i] = (i == 0) ? 0 : sh[i - 1];
    if (threadIdx.x == 0) bsums[blockIdx.x] = sh[SCAN_CHUNK - 1];
}

__global__ void k_scan2(int* bsums, int nb) {     // single block, nb <= 1024
    __shared__ int sh[1024];
    int t = threadIdx.x;
    sh[t] = (t < nb) ? bsums[t] : 0;
    __syncthreads();
    for (int off = 1; off < 1024; off <<= 1) {
        int v = (t >= off) ? sh[t - off] : 0;
        __syncthreads();
        sh[t] += v;
        __syncthreads();
    }
    if (t < nb) bsums[t] = (t == 0) ? 0 : sh[t - 1];
}

__global__ void k_scan3(int* __restrict__ out, const int* __restrict__ bsums, int n) {
    int i = blockIdx.x * blockDim.x + threadIdx.x;
    if (i < n) out[i] += bsums[i / SCAN_CHUNK];
}

__global__ void k_place(const int* __restrict__ row, const int* __restrict__ col,
                        int* __restrict__ ccur, int* __restrict__ erow, int E) {
    int e = blockIdx.x * blockDim.x + threadIdx.x;
    if (e < E) {
        int p = atomicAdd(&ccur[col[e]], 1);
        erow[p] = row[e];
    }
}

// ---------------------------------------------------------------------------
// Node-centric GAT aggregation, wave per target node c.  Zero atomics.
// Pass 1: t_e = inner(x_i, leaky(x_j)) * sigmoid(dinv[r]*inner); store t, max.
// Pass 2: fused exp-sum + weighted accumulate; direct store of agg row.
// ---------------------------------------------------------------------------
__global__ void k_node_gat(const int* __restrict__ coff, const int* __restrict__ erow,
                           const float* __restrict__ xw, const float* __restrict__ dinv,
                           float* __restrict__ tbuf, float* __restrict__ agg, int Nn)
{
    int lane = threadIdx.x & 31;
    int c = blockIdx.x * (blockDim.x >> 5) + (threadIdx.x >> 5);
    if (c >= Nn) return;
    int start = coff[c], end = coff[c + 1];

    float2 xi = ((const float2*)(xw + (long)c * 64))[lane];
    float mx = -__builtin_inff();

    for (int p = start; p < end; ++p) {
        int r = erow[p];
        float2 xj = ((const float2*)(xw + (long)r * 64))[lane];
        float t = xi.x * leakyf(xj.x) + xi.y * leakyf(xj.y);
        #pragma unroll
        for (int off = 16; off > 0; off >>= 1) t += __shfl_xor(t, off, 32);
        float gate = sigf(dinv[r] * t);
        t *= gate;
        if (lane == 0) tbuf[p] = t;
        mx = fmaxf(mx, t);
    }

    // make lane-0's tbuf stores visible to all lanes of this wave
    asm volatile("s_wait_storecnt 0x0" ::: "memory");

    float s = 0.0f;
    float ax = 0.0f, ay = 0.0f;
    for (int p = start; p < end; ++p) {
        int r = erow[p];
        float w = __expf(tbuf[p] - mx);
        s += w;
        float2 xj = ((const float2*)(xw + (long)r * 64))[lane];
        ax += w * xj.x;
        ay += w * xj.y;
    }
    float inv = 1.0f / (s + 1e-16f);
    float* dst = agg + (long)c * 64 + lane * 2;
    dst[0] = ax * inv;
    dst[1] = ay * inv;
}

// ---------------------------------------------------------------------------
extern "C" void kernel_launch(void* const* d_in, const int* in_sizes, int n_in,
                              void* d_out, int out_size, void* d_ws, size_t ws_size,
                              hipStream_t stream) {
    const float* features = (const float*)d_in[0];
    const int*   eidx     = (const int*)d_in[1];
    const float* id_emb   = (const float*)d_in[2];
    const float* pref     = (const float*)d_in[3];
    const float* mlp_w    = (const float*)d_in[4];
    const float* mlp_b    = (const float*)d_in[5];

    const int NI = in_sizes[0] / 512;
    const int E  = in_sizes[1] / 2;
    const int N  = in_sizes[2] / 64;
    const int NU = in_sizes[3] / 64;
    const int* rowp = eidx;
    const int* colp = eidx + E;

    // ---- workspace carve-up ----
    float* x0   = (float*)d_ws;            // [N,64]
    float* xw   = x0   + (long)N * 64;     // [N,64] x@W, reused as h
    float* agg  = xw   + (long)N * 64;     // [N,64] aggregation, reused as x_hat
    float* dinv = agg  + (long)N * 64;     // [N]
    float* tbuf = dinv + N;                // [E]
    uintptr_t wp_addr = (uintptr_t)(tbuf + E);
    wp_addr = (wp_addr + 63) & ~(uintptr_t)63;          // 64B align for v16h loads
    _Float16* wpk = (_Float16*)wp_addr;                 // packed weights
    const long WPK_MLP = (long)512 * 64;                // mlp fragments
    const long WPK_L   = (long)64 * 64;                 // each 64x64 matrix
    _Float16* wpk_end = wpk + WPK_MLP + 9 * WPK_L;
    uintptr_t ip_addr = ((uintptr_t)wpk_end + 63) & ~(uintptr_t)63;
    int* ccnt  = (int*)ip_addr;            // [N]
    int* coff  = ccnt + N;                 // [N+1]
    int* ccur  = coff + N + 1;             // [N]
    int* erow  = ccur + N;                 // [E]
    int* bsums = erow + E;                 // [<=1024]
    float* out = (float*)d_out;            // [N,192]

    const dim3 B(256);
    const int gN  = (N + 255) / 256;
    const int gE  = (E + 255) / 256;
    const int gNw = (N + 7) / 8;                 // wave-per-row/node
    const int gGemm    = (N / 16 + 7) / 8;
    const int gGemmMlp = (NI / 16 + 7) / 8;
    const int nbScan = (N + SCAN_CHUNK - 1) / SCAN_CHUNK;

    // ---- CSR by col (built once, reused by all 3 layers) ----
    k_zeroi<<<gN, B, 0, stream>>>(ccnt, N);
    k_count_col<<<gE, B, 0, stream>>>(colp, ccnt, E);
    k_scan1<<<nbScan, B, 0, stream>>>(ccnt, coff, bsums, N);
    k_scan2<<<1, dim3(1024), 0, stream>>>(bsums, nbScan);
    k_scan3<<<gN, B, 0, stream>>>(coff, bsums, N);
    k_seti<<<1, 1, 0, stream>>>(coff + N, E);
    hipMemcpyAsync(ccur, coff, (size_t)N * sizeof(int), hipMemcpyDeviceToDevice, stream);
    k_place<<<gE, B, 0, stream>>>(rowp, colp, ccur, erow, E);

    // ---- deg^{-1/2} over source index (row) ----
    k_fillf<<<gN, B, 0, stream>>>(dinv, 0.0f, N);
    k_deg<<<gE, B, 0, stream>>>(rowp, dinv, E);
    k_dinv<<<gN, B, 0, stream>>>(dinv, N);

    // ---- pack all weights to f16 WMMA-B fragments ----
    k_pack_w<<<(512 * 64 + 255) / 256, B, 0, stream>>>(mlp_w, 512, wpk);
    for (int i = 0; i < 9; ++i) {
        const float* W = (const float*)d_in[6 + 2 * i];   // conv1,lin1,g1,conv2,...
        k_pack_w<<<(64 * 64 + 255) / 256, B, 0, stream>>>(W, 64, wpk + WPK_MLP + i * WPK_L);
    }

    // ---- x = l2norm(concat(preference, tanh(features@mlp_w + mlp_b))) ----
    hipMemcpyAsync(x0, pref, (size_t)NU * 64 * sizeof(float),
                   hipMemcpyDeviceToDevice, stream);
    k_gemm_wmma<<<gGemmMlp, B, 0, stream>>>(features, 512, 512, NI, wpk, mlp_b,
                                            nullptr, 0, /*tanh*/2, x0 + (long)NU * 64, 64);
    k_rownorm<<<gNw, B, 0, stream>>>(x0, 64, x0, 64, nullptr, 0, N);

    for (int l = 0; l < 3; ++l) {
        const _Float16* conv_wp = wpk + WPK_MLP + (3 * l + 0) * WPK_L;
        const _Float16* lin_wp  = wpk + WPK_MLP + (3 * l + 1) * WPK_L;
        const _Float16* g_wp    = wpk + WPK_MLP + (3 * l + 2) * WPK_L;
        const float* conv_b = (const float*)d_in[6 + 6 * l + 1];
        const float* lin_b  = (const float*)d_in[6 + 6 * l + 3];
        const float* g_b    = (const float*)d_in[6 + 6 * l + 5];
        const float* xin = (l == 0) ? x0 : out + (l - 1) * 64;
        const int    ldx = (l == 0) ? 64 : 192;

        // xW = x @ conv_w
        k_gemm_wmma<<<gGemm, B, 0, stream>>>(xin, ldx, 64, N, conv_wp, nullptr,
                                             nullptr, 0, 0, xw, 64);
        // node-centric attention + aggregation (no atomics)
        k_node_gat<<<gNw, B, 0, stream>>>(coff, erow, xw, dinv, tbuf, agg, N);
        // h = leaky(l2norm(agg + conv_b))  -> xw
        k_rownorm<<<gNw, B, 0, stream>>>(agg, 64, xw, 64, conv_b, 1, N);
        // x_hat = leaky(x@lin_w + lin_b) + id_embedding  -> agg
        k_gemm_wmma<<<gGemm, B, 0, stream>>>(xin, ldx, 64, N, lin_wp, lin_b,
                                             id_emb, /*after*/0, 1, agg, 64);
        // x_{l+1} = leaky(h@g_w + g_b + x_hat) -> d_out column block l (ldc=192)
        k_gemm_wmma<<<gGemm, B, 0, stream>>>(xw, 64, 64, N, g_wp, g_b,
                                             agg, /*before*/1, 1, out + l * 64, 192);
    }
}
